// MultiHeadSelfAttention_51754355917453
// MI455X (gfx1250) — matrix operand order repaired and checked by reading, not compiled, not run on hardware
//
#include <hip/hip_runtime.h>

#define D_MODEL 1024
#define S_LEN   2048
#define BATCH   2
#define NHEADS  16
#define DK      64
#define NQKV    3072

typedef __attribute__((ext_vector_type(16))) __bf16 v16bf;
typedef __attribute__((ext_vector_type(8)))  float  v8f;

union Frag16 { uint4 q[2]; v16bf v; unsigned short s[16]; };

// Native f32 -> bf16 conversion (backend lowers to v_cvt_*bf16*), stored as u16.
__device__ __forceinline__ unsigned short bfc(float f) {
  __bf16 h = (__bf16)f;
  return __builtin_bit_cast(unsigned short, h);
}

__device__ __forceinline__ v8f wmma_bf16(v16bf a, v16bf b, v8f c) {
  // D = A(16x32 bf16) x B(32x16 bf16) + C(16x16 f32)
  return __builtin_amdgcn_wmma_f32_16x16x32_bf16(false, a, false, b,
                                                 (short)0, c, false, false);
}

// A-fragment pair (rows m0..m0+31) at depth k0
__device__ __forceinline__ void load_a2(Frag16* a,
                                        const unsigned short* xrow0,
                                        const unsigned short* xrow1,
                                        int k0, int half) {
  const unsigned short* p0 = xrow0 + k0 + half * 16;
  a[0].q[0] = *(const uint4*)(p0);
  a[0].q[1] = *(const uint4*)(p0 + 8);
  const unsigned short* p1 = xrow1 + k0 + half * 16;
  a[1].q[0] = *(const uint4*)(p1);
  a[1].q[1] = *(const uint4*)(p1 + 8);
}

// 4 B-fragments (64 output features) at depth k0 ; wrow[t] pre-offset by half*16
__device__ __forceinline__ void load_b4(Frag16* b,
                                        const unsigned short* const* wrow,
                                        int k0) {
#pragma unroll
  for (int t = 0; t < 4; ++t) {
    const unsigned short* p = wrow[t] + k0;
    b[t].q[0] = *(const uint4*)(p);
    b[t].q[1] = *(const uint4*)(p + 8);
  }
}

__device__ __forceinline__ void mma8(v8f acc[2][4], const Frag16* a, const Frag16* b) {
#pragma unroll
  for (int t = 0; t < 4; ++t)
#pragma unroll
    for (int i = 0; i < 2; ++i)
      acc[i][t] = wmma_bf16(a[i].v, b[t].v, acc[i][t]);
}

// ---------------------------------------------------------------------------
// Kernel 0: one-time fp32 -> bf16 down-convert (8 elements/thread, b128 I/O).
// ---------------------------------------------------------------------------
__global__ __launch_bounds__(256)
void cvt_bf16_kernel(const float* __restrict__ src,
                     unsigned short* __restrict__ dst, int n)
{
  const int i = (blockIdx.x * 256 + threadIdx.x) * 8;
  if (i >= n) return;
  float4 a = *(const float4*)(src + i);
  float4 b = *(const float4*)(src + i + 4);
  uint4 pk;
  pk.x = (unsigned)bfc(a.x) | ((unsigned)bfc(a.y) << 16);
  pk.y = (unsigned)bfc(a.z) | ((unsigned)bfc(a.w) << 16);
  pk.z = (unsigned)bfc(b.x) | ((unsigned)bfc(b.y) << 16);
  pk.w = (unsigned)bfc(b.z) | ((unsigned)bfc(b.w) << 16);
  *(uint4*)(dst + i) = pk;
}

// ---------------------------------------------------------------------------
// Kernel 1: qkv = x @ w_qkv^T (bf16 operands, f32 accum), fused RoPE on q/k.
//   Q,K  -> [B,H,S,dk] ; V^T -> [B,H,dk,S]
// wave tile: 32(M) x 64(N), double-buffered k-loop, 8 WMMA per 32-deep step.
// ---------------------------------------------------------------------------
__global__ __launch_bounds__(128)
void qkv_rope_kernel(const unsigned short* __restrict__ xbf,
                     const int* __restrict__ tpos,
                     const unsigned short* __restrict__ wbf,
                     unsigned short* __restrict__ Qw,
                     unsigned short* __restrict__ Kw,
                     unsigned short* __restrict__ Vtw)
{
  const int lane = threadIdx.x & 31;
  const int wave = threadIdx.x >> 5;
  const int col  = lane & 15;
  const int half = lane >> 4;
  const int tid  = blockIdx.x * 4 + wave;          // 6144 wave tiles
  const int NT   = NQKV / 64;                      // 48
  const int mt   = tid / NT;
  const int nt   = tid - mt * NT;
  const int m0   = mt << 5;                        // 32-row tile
  const int n0   = nt << 6;

  v8f acc[2][4];
#pragma unroll
  for (int i = 0; i < 2; ++i)
#pragma unroll
    for (int t = 0; t < 4; ++t)
#pragma unroll
      for (int r = 0; r < 8; ++r) acc[i][t][r] = 0.0f;

  const unsigned short* xrow0 = xbf + (size_t)(m0 + col) * D_MODEL;
  const unsigned short* xrow1 = xbf + (size_t)(m0 + 16 + col) * D_MODEL;
  const unsigned short* wrow[4];
#pragma unroll
  for (int t = 0; t < 4; ++t)
    wrow[t] = wbf + (size_t)(n0 + 16 * t + col) * D_MODEL + half * 16;

  Frag16 aP[2], bP[4], aQ[2], bQ[4];
  load_a2(aP, xrow0, xrow1, 0, half);
  load_b4(bP, wrow, 0);
  for (int k0 = 0; k0 < D_MODEL; k0 += 64) {       // 2-stage pipeline
    load_a2(aQ, xrow0, xrow1, k0 + 32, half);
    load_b4(bQ, wrow, k0 + 32);
    mma8(acc, aP, bP);
    if (k0 + 64 < D_MODEL) {
      load_a2(aP, xrow0, xrow1, k0 + 64, half);
      load_b4(bP, wrow, k0 + 64);
    }
    mma8(acc, aQ, bQ);
  }

  const int sec = n0 >> 10;                        // 0=q,1=k,2=v
  const int h   = (n0 >> 6) & 15;

  if (sec == 2) {                                  // V: store transposed [B,H,dk,S]
#pragma unroll
    for (int i = 0; i < 2; ++i)
#pragma unroll
      for (int t = 0; t < 4; ++t) {
        const int idx = 16 * t + col;
        const int m   = m0 + 16 * i + half * 8;
        const int b   = m >> 11;
        const int sb  = m & (S_LEN - 1);
        uint4 pk;
        pk.x = (unsigned)bfc(acc[i][t][0]) | ((unsigned)bfc(acc[i][t][1]) << 16);
        pk.y = (unsigned)bfc(acc[i][t][2]) | ((unsigned)bfc(acc[i][t][3]) << 16);
        pk.z = (unsigned)bfc(acc[i][t][4]) | ((unsigned)bfc(acc[i][t][5]) << 16);
        pk.w = (unsigned)bfc(acc[i][t][6]) | ((unsigned)bfc(acc[i][t][7]) << 16);
        *(uint4*)(Vtw + (((size_t)(b * NHEADS + h) * DK + idx) * S_LEN + sb)) = pk;
      }
  } else {                                         // Q/K: RoPE then store [B,H,S,dk]
    unsigned short* dstbase = sec ? Kw : Qw;
#pragma unroll
    for (int t = 0; t < 4; ++t) {
      const int   idx  = 16 * t + col;
      const float freq = exp2f(-13.287712379549449f * (float)(idx & ~1) * (1.0f / 64.0f));
#pragma unroll
      for (int i = 0; i < 2; ++i)
#pragma unroll
        for (int r = 0; r < 8; ++r) {
          const int   m    = m0 + 16 * i + half * 8 + r;
          const int   b    = m >> 11;
          const int   srow = m & (S_LEN - 1);
          const float pos  = (float)tpos[srow];
          const float ang  = pos * freq;
          const float cs   = __cosf(ang);
          const float sn   = __sinf(ang);
          const float self = acc[i][t][r];
          const float part = __shfl_xor(self, 1, 32); // pair (even,odd) adjacent lanes
          const float v = (idx & 1) ? (part * sn + self * cs)
                                    : (self * cs - part * sn);
          dstbase[((size_t)(b * NHEADS + h) * S_LEN + srow) * DK + idx] = bfc(v);
        }
    }
  }
}

// ---------------------------------------------------------------------------
// Kernel 2: causal flash attention, one wave per 16 q-rows of one (b,h).
// Scores computed transposed (S^T = K x Q^T) so the two 16x16 score tiles of a
// 32-wide k step ARE the A-fragment of P for the P*V WMMA (no transpose).
// ---------------------------------------------------------------------------
__global__ __launch_bounds__(128)
void attn_kernel(const unsigned short* __restrict__ Q,
                 const unsigned short* __restrict__ K,
                 const unsigned short* __restrict__ Vt,
                 unsigned short* __restrict__ AO)
{
  const int lane = threadIdx.x & 31;
  const int wave = threadIdx.x >> 5;
  const int col  = lane & 15;
  const int half = lane >> 4;
  const int tile = blockIdx.x * 4 + wave;          // 4096 tiles
  const int bh   = tile >> 7;                      // / (S/16)
  const int qt   = tile & 127;
  const int q0   = qt << 4;
  const int b    = bh >> 4;
  const int h    = bh & 15;

  const unsigned short* Qb = Q  + (size_t)bh * S_LEN * DK;
  const unsigned short* Kb = K  + (size_t)bh * S_LEN * DK;
  const unsigned short* Vb = Vt + (size_t)bh * DK * S_LEN;

  Frag16 qf[2];                                    // B-frags of Q^T, held all loop
#pragma unroll
  for (int c = 0; c < 2; ++c) {
    const unsigned short* p = Qb + (size_t)(q0 + col) * DK + c * 32 + half * 16;
    qf[c].q[0] = *(const uint4*)(p);
    qf[c].q[1] = *(const uint4*)(p + 8);
  }

  float m_run = -1e30f, l_run = 0.0f;
  v8f acc[4];
#pragma unroll
  for (int t = 0; t < 4; ++t)
#pragma unroll
    for (int r = 0; r < 8; ++r) acc[t][r] = 0.0f;

  const float L2E = 1.4426950408889634f;
  const int kend = q0 + 16;                        // wave-uniform -> EXEC all ones
  for (int k0 = 0; k0 < kend; k0 += 32) {
    v8f st[2];
#pragma unroll
    for (int kk = 0; kk < 2; ++kk) {
#pragma unroll
      for (int r = 0; r < 8; ++r) st[kk][r] = 0.0f;
#pragma unroll
      for (int c = 0; c < 2; ++c) {                // A-frag of K tile (rows = k)
        Frag16 kf;
        const unsigned short* p =
            Kb + (size_t)(k0 + 16 * kk + col) * DK + c * 32 + half * 16;
        kf.q[0] = *(const uint4*)(p);
        kf.q[1] = *(const uint4*)(p + 8);
        st[kk] = wmma_bf16(kf.v, qf[c].v, st[kk]); // S^T[k,q]
      }
    }
    // scale + causal mask (q = q0+col ; k = k0+16kk+half*8+r)
    float mx = -1e30f;
#pragma unroll
    for (int kk = 0; kk < 2; ++kk)
#pragma unroll
      for (int r = 0; r < 8; ++r) {
        const int krow = k0 + 16 * kk + half * 8 + r;
        float v = st[kk][r] * 0.125f;              // 1/sqrt(64)
        if (krow > q0 + col) v = -1e30f;
        st[kk][r] = v;
        mx = fmaxf(mx, v);
      }
    mx = fmaxf(mx, __shfl_xor(mx, 16, 32));
    const float m_new = fmaxf(m_run, mx);

    Frag16 pa;                                     // A-frag of P (16q x 32k), free
    float psum = 0.0f;
#pragma unroll
    for (int kk = 0; kk < 2; ++kk)
#pragma unroll
      for (int r = 0; r < 8; ++r) {
        const float p = exp2f((st[kk][r] - m_new) * L2E);
        psum += p;
        pa.s[kk * 8 + r] = bfc(p);
      }
    psum += __shfl_xor(psum, 16, 32);
    const float fac = exp2f((m_run - m_new) * L2E);
    l_run = l_run * fac + psum;
    m_run = m_new;

    float facs[8];                                 // per-q-row rescale, q stats live in lane==q
#pragma unroll
    for (int r = 0; r < 8; ++r) facs[r] = __shfl(fac, half * 8 + r, 32);
#pragma unroll
    for (int t = 0; t < 4; ++t) {
#pragma unroll
      for (int r = 0; r < 8; ++r) acc[t][r] *= facs[r];
      Frag16 vf;                                   // B-frag of V from V^T: table order
      const unsigned short* p = Vb + (size_t)(16 * t + col) * S_LEN + k0 + half * 8;
      vf.q[0] = *(const uint4*)(p);
      vf.q[1] = *(const uint4*)(p + 16);
      acc[t] = wmma_bf16(pa.v, vf.v, acc[t]);      // out[q,d]
    }
  }

  float ls[8];
#pragma unroll
  for (int r = 0; r < 8; ++r) ls[r] = __shfl(l_run, half * 8 + r, 32);

  unsigned short* ob = AO + (size_t)b * S_LEN * D_MODEL;
#pragma unroll
  for (int t = 0; t < 4; ++t)
#pragma unroll
    for (int r = 0; r < 8; ++r) {
      const int srow = q0 + half * 8 + r;
      const float o = acc[t][r] / ls[r];
      ob[(size_t)srow * D_MODEL + h * DK + 16 * t + col] = bfc(o);
    }
}

// ---------------------------------------------------------------------------
// Kernel 3: out = attn_out(bf16) @ w_out^T(bf16) -> fp32 ; 32x64 wave tiles.
// ---------------------------------------------------------------------------
__global__ __launch_bounds__(128)
void out_proj_kernel(const unsigned short* __restrict__ A,
                     const unsigned short* __restrict__ wbf,
                     float* __restrict__ out)
{
  const int lane = threadIdx.x & 31;
  const int wave = threadIdx.x >> 5;
  const int col  = lane & 15;
  const int half = lane >> 4;
  const int tid  = blockIdx.x * 4 + wave;          // 2048 wave tiles
  const int NT   = D_MODEL / 64;                   // 16
  const int mt   = tid / NT;
  const int nt   = tid - mt * NT;
  const int m0   = mt << 5;
  const int n0   = nt << 6;

  v8f acc[2][4];
#pragma unroll
  for (int i = 0; i < 2; ++i)
#pragma unroll
    for (int t = 0; t < 4; ++t)
#pragma unroll
      for (int r = 0; r < 8; ++r) acc[i][t][r] = 0.0f;

  const unsigned short* arow0 = A + (size_t)(m0 + col) * D_MODEL;
  const unsigned short* arow1 = A + (size_t)(m0 + 16 + col) * D_MODEL;
  const unsigned short* wrow[4];
#pragma unroll
  for (int t = 0; t < 4; ++t)
    wrow[t] = wbf + (size_t)(n0 + 16 * t + col) * D_MODEL + half * 16;

  Frag16 aP[2], bP[4], aQ[2], bQ[4];
  load_a2(aP, arow0, arow1, 0, half);
  load_b4(bP, wrow, 0);
  for (int k0 = 0; k0 < D_MODEL; k0 += 64) {       // 2-stage pipeline
    load_a2(aQ, arow0, arow1, k0 + 32, half);
    load_b4(bQ, wrow, k0 + 32);
    mma8(acc, aP, bP);
    if (k0 + 64 < D_MODEL) {
      load_a2(aP, arow0, arow1, k0 + 64, half);
      load_b4(bP, wrow, k0 + 64);
    }
    mma8(acc, aQ, bQ);
  }

#pragma unroll
  for (int i = 0; i < 2; ++i)
#pragma unroll
    for (int t = 0; t < 4; ++t)
#pragma unroll
      for (int r = 0; r < 8; ++r) {
        const int m = m0 + 16 * i + half * 8 + r;
        out[(size_t)m * D_MODEL + n0 + 16 * t + col] = acc[i][t][r];
      }
}

extern "C" void kernel_launch(void* const* d_in, const int* in_sizes, int n_in,
                              void* d_out, int out_size, void* d_ws, size_t ws_size,
                              hipStream_t stream) {
  (void)in_sizes; (void)n_in; (void)out_size; (void)ws_size;
  const float* x     = (const float*)d_in[0];
  const int*   tpos  = (const int*)d_in[1];
  const float* w_qkv = (const float*)d_in[2];
  const float* w_out = (const float*)d_in[3];
  float*       out   = (float*)d_out;

  const size_t NBHSD = (size_t)BATCH * NHEADS * S_LEN * DK;   // 4M elems
  const int    NX    = BATCH * S_LEN * D_MODEL;               // 4,194,304
  const int    NWQKV = NQKV * D_MODEL;                        // 3,145,728
  const int    NWOUT = D_MODEL * D_MODEL;                     // 1,048,576

  unsigned short* Qw   = (unsigned short*)d_ws;
  unsigned short* Kw   = Qw   + NBHSD;
  unsigned short* Vtw  = Kw   + NBHSD;
  unsigned short* AOw  = Vtw  + NBHSD;
  unsigned short* Xbf  = AOw  + NBHSD;
  unsigned short* Wqbf = Xbf  + NX;
  unsigned short* Wobf = Wqbf + NWQKV;                        // 48 MB total

  // One-time down-converts (HBM-bound, ~1.5 us total at 23.3 TB/s)
  cvt_bf16_kernel<<<NX / 2048, 256, 0, stream>>>(x, Xbf, NX);
  cvt_bf16_kernel<<<NWQKV / 2048, 256, 0, stream>>>(w_qkv, Wqbf, NWQKV);
  cvt_bf16_kernel<<<NWOUT / 2048, 256, 0, stream>>>(w_out, Wobf, NWOUT);

  // 128 m-tiles(32) * 48 n-tiles = 6144 wave tiles / 4 waves
  qkv_rope_kernel<<<1536, 128, 0, stream>>>(Xbf, tpos, Wqbf, Qw, Kw, Vtw);
  // 2*16*128 = 4096 wave tiles / 4 waves
  attn_kernel<<<1024, 128, 0, stream>>>(Qw, Kw, Vtw, AOw);
  // 128 m-tiles(32) * 16 n-tiles = 2048 wave tiles / 4 waves
  out_proj_kernel<<<512, 128, 0, stream>>>(AOw, Wobf, out);
}
